// MultiHeadAttention_56418690400535
// MI455X (gfx1250) — compile-verified, matrix operand order checked by reading, not run on hardware
//
#include <hip/hip_runtime.h>
#include <math.h>

// ---------------------------------------------------------------------------
// MHA forward for MI455X (gfx1250, wave32, WMMA bf16 16x16x32, f32 accum).
// All GEMM operands stored k-contiguous so every fragment load coalesces to
// global_load_b128; each wave computes a 16x64 strip (1 A-frag : 4 WMMAs).
// ---------------------------------------------------------------------------

typedef __bf16 bf16;
typedef __attribute__((ext_vector_type(16))) __bf16 v16bf;
typedef __attribute__((ext_vector_type(8)))  float  v8f;

#define Bn 4
#define Tn 1024
#define Dn 1024
#define Hn 16
#define HSn 64
#define SCALE 0.03125f   // 1024^-0.5

// ---- fp32 -> bf16 round-to-nearest-even (pure integer)
__device__ __forceinline__ bf16 f2bf(float f) {
  unsigned u = __builtin_bit_cast(unsigned, f);
  u += 0x7FFFu + ((u >> 16) & 1u);
  unsigned short h = (unsigned short)(u >> 16);
  return __builtin_bit_cast(bf16, h);
}

// K-pair start for VGPR j, lane-half (ISA 16-bit A 16x32 layout).
// For half h: j=0..3 -> k = 2j + 8h (contiguous 0..7 / 8..15),
//             j=4..7 -> k = 16 + 2(j-4) + 8h (contiguous 16..23 / 24..31).
__device__ __forceinline__ int kpair(int j, int half) {
  return 2 * j + ((j >= 4) ? 8 : 0) + 8 * half;
}

__device__ __forceinline__ v8f wmma_bf16(v16bf a, v16bf b, v8f c) {
  return __builtin_amdgcn_wmma_f32_16x16x32_bf16(
      /*neg_a=*/false, a, /*neg_b=*/false, b,
      /*c_mod=*/(short)0, c, /*reuse_a=*/false, /*reuse_b=*/false);
}

// Fragment load, k-contiguous memory: element (m, k) at A[m*ld + k].
// Used for A fragments AND for B fragments of pre-transposed matrices
// (B value (k, n) stored at Bt[n*ld + k]).  Lowers to 2x global_load_b128.
__device__ __forceinline__ v16bf load_frag_bf16(const bf16* __restrict__ A,
                                                int row0, int k0, int ld) {
  const int lane = threadIdx.x & 31;
  const int m = lane & 15, half = lane >> 4;
  const bf16* p = A + (size_t)(row0 + m) * ld + k0;
  v16bf a;
#pragma unroll
  for (int j = 0; j < 8; ++j) {
    const int kk = kpair(j, half);
    a[2 * j]     = p[kk];
    a[2 * j + 1] = p[kk + 1];
  }
  return a;
}

// Same pattern from fp32 memory, converted on load (chunks coalesce to b128).
__device__ __forceinline__ v16bf load_frag_f32(const float* __restrict__ A,
                                               int row0, int k0, int ld) {
  const int lane = threadIdx.x & 31;
  const int m = lane & 15, half = lane >> 4;
  const float* p = A + (size_t)(row0 + m) * ld + k0;
  v16bf a;
#pragma unroll
  for (int j = 0; j < 8; ++j) {
    const int kk = kpair(j, half);
    a[2 * j]     = f2bf(p[kk]);
    a[2 * j + 1] = f2bf(p[kk + 1]);
  }
  return a;
}

// ---------------------------------------------------------------------------
// Kernel 1a: fp32 -> bf16 convert (x)
// ---------------------------------------------------------------------------
__global__ void cvt_f32_bf16(const float* __restrict__ s, bf16* __restrict__ d,
                             int n) {
  int i = blockIdx.x * blockDim.x + threadIdx.x;
  if (i < n) d[i] = f2bf(s[i]);
}

// ---------------------------------------------------------------------------
// Kernel 1b: fp32 [nmat][K][N] -> bf16 [nmat][N][K] transpose-convert.
// dst indexed linearly (coalesced writes), src strided reads (L2-served).
// ---------------------------------------------------------------------------
__global__ void cvt_transpose(const float* __restrict__ s,
                              bf16* __restrict__ d, int K, int N, int nmat) {
  size_t i = (size_t)blockIdx.x * blockDim.x + threadIdx.x;
  const size_t per = (size_t)K * N;
  if (i >= per * (size_t)nmat) return;
  const int m = (int)(i / per);
  const size_t r = i % per;
  const int n = (int)(r / K), k = (int)(r % K);
  d[i] = f2bf(s[(size_t)m * per + (size_t)k * N + n]);
}

// ---------------------------------------------------------------------------
// Kernel 2: QKV projections.  One wave = one 16(t) x 64(e) strip.
// Wt layout [H][HS][D] (k-contiguous).  K,Q stored [B,H,T,HS];
// V stored TRANSPOSED [B,H,HS,T] (k-contiguous for the attn@V GEMM).
// waves = 3 * B * H * (T/16) = 12288
// ---------------------------------------------------------------------------
__global__ void proj_qkv(const bf16* __restrict__ xbf,
                         const bf16* __restrict__ Wkt,
                         const bf16* __restrict__ Wqt,
                         const bf16* __restrict__ Wvt,
                         const float* __restrict__ bk,
                         const float* __restrict__ bq,
                         const float* __restrict__ bv,
                         bf16* __restrict__ Kb, bf16* __restrict__ Qb,
                         bf16* __restrict__ Vt) {
  int tile = blockIdx.x * (blockDim.x >> 5) + (threadIdx.x >> 5);
  const int tm = tile & 63; tile >>= 6;          // T/16 = 64
  const int h  = tile & 15; tile >>= 4;          // H
  const int b  = tile & 3;  tile >>= 2;          // B
  const int mat = tile;                          // 0=K 1=Q 2=V
  if (mat >= 3) return;

  const bf16*  Wt   = (mat == 0) ? Wkt : (mat == 1) ? Wqt : Wvt;
  const float* bias = (mat == 0) ? bk  : (mat == 1) ? bq  : bv;

  const bf16* Ab = xbf + (size_t)b * Tn * Dn;          // [T][D]
  const bf16* Bt = Wt + (size_t)h * HSn * Dn;          // [HS][D]

  v8f acc[4] = {};
#pragma unroll 2
  for (int k0 = 0; k0 < Dn; k0 += 32) {
    v16bf a = load_frag_bf16(Ab, tm * 16, k0, Dn);
#pragma unroll
    for (int c = 0; c < 4; ++c) {
      v16bf bb = load_frag_bf16(Bt, c * 16, k0, Dn);
      acc[c] = wmma_bf16(a, bb, acc[c]);
    }
  }

  const int lane = threadIdx.x & 31;
  const int n = lane & 15, half = lane >> 4;
  if (mat != 2) {
    bf16* Op = ((mat == 0) ? Kb : Qb) + (((size_t)b * Hn + h) * Tn) * HSn;
#pragma unroll
    for (int c = 0; c < 4; ++c) {
      const int col = c * 16 + n;
      const float bv_ = bias[h * HSn + col];
#pragma unroll
      for (int r = 0; r < 8; ++r) {
        const int row = tm * 16 + r + 8 * half;
        Op[(size_t)row * HSn + col] = f2bf(acc[c][r] + bv_);
      }
    }
  } else {  // V transposed: Vt[b][h][e][t]
    bf16* Op = Vt + (((size_t)b * Hn + h) * HSn) * Tn;
#pragma unroll
    for (int c = 0; c < 4; ++c) {
      const int col = c * 16 + n;                 // e
      const float bv_ = bias[h * HSn + col];
#pragma unroll
      for (int r = 0; r < 8; ++r) {
        const int row = tm * 16 + r + 8 * half;   // t
        Op[(size_t)col * Tn + row] = f2bf(acc[c][r] + bv_);
      }
    }
  }
}

// ---------------------------------------------------------------------------
// Kernel 3: scores S[b,h,t,s] = scale * (K[t,:] . Q[s,:]), causal (s<=t).
// One wave = 16(t) x 64(s) strip.  waves = B * H * 64 * 16 = 65536
// ---------------------------------------------------------------------------
__global__ void scores_kernel(const bf16* __restrict__ Kb,
                              const bf16* __restrict__ Qb,
                              float* __restrict__ attn) {
  int tile = blockIdx.x * (blockDim.x >> 5) + (threadIdx.x >> 5);
  const int ts = tile & 15; tile >>= 4;          // s-strip (64 wide)
  const int tt = tile & 63; tile >>= 6;          // t-tile (16)
  const int h  = tile & 15; tile >>= 4;
  const int b  = tile;
  if (b >= Bn) return;

  const int t0 = tt * 16, s0 = ts * 64;
  float* S = attn + (((size_t)b * Hn + h) * Tn) * Tn;

  const int lane = threadIdx.x & 31;
  const int n = lane & 15, half = lane >> 4;
  const float ninf = -__builtin_inff();

  if (s0 > t0 + 15) {  // wave-uniform: whole strip above the diagonal
#pragma unroll
    for (int c = 0; c < 4; ++c)
#pragma unroll
      for (int r = 0; r < 8; ++r)
        S[(size_t)(t0 + r + 8 * half) * Tn + s0 + c * 16 + n] = ninf;
    return;
  }

  const bf16* Kp = Kb + (((size_t)b * Hn + h) * Tn) * HSn;
  const bf16* Qp = Qb + (((size_t)b * Hn + h) * Tn) * HSn;

  v8f acc[4] = {};
#pragma unroll
  for (int k0 = 0; k0 < HSn; k0 += 32) {
    v16bf a = load_frag_bf16(Kp, t0, k0, HSn);           // A(t, e)
#pragma unroll
    for (int c = 0; c < 4; ++c) {
      v16bf bb = load_frag_bf16(Qp, s0 + c * 16, k0, HSn);  // B(e,s)=Q[s,e]
      acc[c] = wmma_bf16(a, bb, acc[c]);
    }
  }

#pragma unroll
  for (int c = 0; c < 4; ++c)
#pragma unroll
    for (int r = 0; r < 8; ++r) {
      const int t = t0 + r + 8 * half;
      const int s = s0 + c * 16 + n;
      S[(size_t)t * Tn + s] = (s <= t) ? acc[c][r] * SCALE : ninf;
    }
}

// ---------------------------------------------------------------------------
// Kernel 4: in-place row softmax over T=1024 (one 256-thread block per row)
// ---------------------------------------------------------------------------
__global__ void softmax_rows(float* __restrict__ S) {
  __shared__ float red[256];
  float* p = S + (size_t)blockIdx.x * Tn;
  const int tid = threadIdx.x;

  float m = -__builtin_inff();
  float vals[Tn / 256];
#pragma unroll
  for (int j = 0; j < Tn / 256; ++j) {
    vals[j] = p[tid + j * 256];
    m = fmaxf(m, vals[j]);
  }
  red[tid] = m;
  __syncthreads();
  for (int s = 128; s > 0; s >>= 1) {
    if (tid < s) red[tid] = fmaxf(red[tid], red[tid + s]);
    __syncthreads();
  }
  m = red[0];
  __syncthreads();

  float sum = 0.f;
#pragma unroll
  for (int j = 0; j < Tn / 256; ++j) {
    vals[j] = __expf(vals[j] - m);   // exp(-inf - m) == 0
    sum += vals[j];
  }
  red[tid] = sum;
  __syncthreads();
  for (int s = 128; s > 0; s >>= 1) {
    if (tid < s) red[tid] += red[tid + s];
    __syncthreads();
  }
  const float inv = 1.0f / red[0];
#pragma unroll
  for (int j = 0; j < Tn / 256; ++j) p[tid + j * 256] = vals[j] * inv;
}

// ---------------------------------------------------------------------------
// Kernel 5: O[b,h,t,e] = sum_s P[t,s] * V[s,e].  V is transposed [B,H,HS,T]
// so B-fragments are k-contiguous.  One wave = 16(t) x 64(e) strip (full HS).
// Output written bf16 into concat layout [B,T,D].  waves = B*H*64 = 4096
// ---------------------------------------------------------------------------
__global__ void attn_v(const float* __restrict__ attn,
                       const bf16* __restrict__ Vt,
                       bf16* __restrict__ ybf) {
  int tile = blockIdx.x * (blockDim.x >> 5) + (threadIdx.x >> 5);
  const int tm = tile & 63; tile >>= 6;
  const int h  = tile & 15; tile >>= 4;
  const int b  = tile;
  if (b >= Bn) return;

  const float* Pp = attn + (((size_t)b * Hn + h) * Tn) * Tn;  // [T][T]
  const bf16*  Vp = Vt + (((size_t)b * Hn + h) * HSn) * Tn;   // [HS][T]

  v8f acc[4] = {};
  for (int k0 = 0; k0 < Tn; k0 += 32) {
    v16bf a = load_frag_f32(Pp, tm * 16, k0, Tn);
#pragma unroll
    for (int c = 0; c < 4; ++c) {
      v16bf bb = load_frag_bf16(Vp, c * 16, k0, Tn);  // B(s,e)=Vt[e][s]
      acc[c] = wmma_bf16(a, bb, acc[c]);
    }
  }

  const int lane = threadIdx.x & 31;
  const int n = lane & 15, half = lane >> 4;
#pragma unroll
  for (int c = 0; c < 4; ++c) {
    const int col = h * HSn + c * 16 + n;       // concat-head feature index
#pragma unroll
    for (int r = 0; r < 8; ++r) {
      const int row = tm * 16 + r + 8 * half;   // t
      ybf[((size_t)b * Tn + row) * Dn + col] = f2bf(acc[c][r]);
    }
  }
}

// ---------------------------------------------------------------------------
// Kernel 6: out = y @ Wp + bp  (M = B*T, N = K = D), Wp pre-transposed [N][K].
// One wave = 16(m) x 64(n) strip.  waves = (B*T/16) * (D/64) = 4096
// ---------------------------------------------------------------------------
__global__ void proj_out(const bf16* __restrict__ ybf,
                         const bf16* __restrict__ Wpt,
                         const float* __restrict__ bp,
                         float* __restrict__ out) {
  int tile = blockIdx.x * (blockDim.x >> 5) + (threadIdx.x >> 5);
  const int tns = tile & 15; tile >>= 4;     // n-strip: D/64 = 16
  const int tm  = tile;                      // (B*T)/16 = 256
  if (tm >= (Bn * Tn) / 16) return;

  v8f acc[4] = {};
#pragma unroll 2
  for (int k0 = 0; k0 < Dn; k0 += 32) {
    v16bf a = load_frag_bf16(ybf, tm * 16, k0, Dn);
#pragma unroll
    for (int c = 0; c < 4; ++c) {
      v16bf bb = load_frag_bf16(Wpt, tns * 64 + c * 16, k0, Dn);
      acc[c] = wmma_bf16(a, bb, acc[c]);
    }
  }

  const int lane = threadIdx.x & 31;
  const int n = lane & 15, half = lane >> 4;
#pragma unroll
  for (int c = 0; c < 4; ++c) {
    const int col = tns * 64 + c * 16 + n;
    const float bias = bp[col];
#pragma unroll
    for (int r = 0; r < 8; ++r) {
      const int row = tm * 16 + r + 8 * half;
      out[(size_t)row * Dn + col] = acc[c][r] + bias;
    }
  }
}

// ---------------------------------------------------------------------------
extern "C" void kernel_launch(void* const* d_in, const int* in_sizes, int n_in,
                              void* d_out, int out_size, void* d_ws,
                              size_t ws_size, hipStream_t stream) {
  const float* x  = (const float*)d_in[0];
  const float* Wk = (const float*)d_in[1];
  const float* bk = (const float*)d_in[2];
  const float* Wq = (const float*)d_in[3];
  const float* bq = (const float*)d_in[4];
  const float* Wv = (const float*)d_in[5];
  const float* bv = (const float*)d_in[6];
  const float* Wp = (const float*)d_in[7];
  const float* bp = (const float*)d_in[8];

  float* outp = (float*)d_out;                          // [B,T,D]
  float* attn = outp + (size_t)Bn * Tn * Dn;            // [B,H,T,T]

  // workspace layout (bf16 elements), total 24M elems = 48 MB
  bf16* xbf  = (bf16*)d_ws;                  // B*T*D      = 4,194,304
  bf16* Wkt  = xbf  + (size_t)Bn * Tn * Dn;  // [H][HS][D] = 1,048,576
  bf16* Wqt  = Wkt  + (size_t)Hn * HSn * Dn;
  bf16* Wvt  = Wqt  + (size_t)Hn * HSn * Dn;
  bf16* Wpt  = Wvt  + (size_t)Hn * HSn * Dn; // [D][D]^T   = 1,048,576
  bf16* Kb   = Wpt  + (size_t)Dn * Dn;       // [B,H,T,HS] = 4,194,304
  bf16* Qb   = Kb   + (size_t)Bn * Hn * Tn * HSn;
  bf16* Vt   = Qb   + (size_t)Bn * Hn * Tn * HSn;       // [B,H,HS,T]
  bf16* ybf  = Vt   + (size_t)Bn * Hn * Tn * HSn;       // [B,T,D]

  const int nX = Bn * Tn * Dn;        // 4,194,304
  const int nW = Hn * Dn * HSn;       // 1,048,576

  // 1) converts (x straight; weights transpose-converted to k-contiguous)
  cvt_f32_bf16<<<(nX + 255) / 256, 256, 0, stream>>>(x, xbf, nX);
  cvt_transpose<<<(nW + 255) / 256, 256, 0, stream>>>(Wk, Wkt, Dn, HSn, Hn);
  cvt_transpose<<<(nW + 255) / 256, 256, 0, stream>>>(Wq, Wqt, Dn, HSn, Hn);
  cvt_transpose<<<(nW + 255) / 256, 256, 0, stream>>>(Wv, Wvt, Dn, HSn, Hn);
  cvt_transpose<<<(nW + 255) / 256, 256, 0, stream>>>(Wp, Wpt, Dn, Dn, 1);

  // 2) QKV projections: 12288 waves / 8 per block
  proj_qkv<<<12288 / 8, 256, 0, stream>>>(xbf, Wkt, Wqt, Wvt, bk, bq, bv, Kb,
                                          Qb, Vt);

  // 3) scores + causal mask: 65536 waves
  scores_kernel<<<65536 / 8, 256, 0, stream>>>(Kb, Qb, attn);

  // 4) softmax: one block per row, B*H*T = 65536 rows
  softmax_rows<<<Bn * Hn * Tn, 256, 0, stream>>>(attn);

  // 5) attn @ V: 4096 waves
  attn_v<<<4096 / 8, 256, 0, stream>>>(attn, Vt, ybf);

  // 6) output projection: 4096 waves
  proj_out<<<4096 / 8, 256, 0, stream>>>(ybf, Wpt, bp, outp);
}
